// AdaptivePoolingAttention_67250597921106
// MI455X (gfx1250) — matrix operand order repaired and checked by reading, not compiled, not run on hardware
//
#include <hip/hip_runtime.h>
#include <hip/hip_bf16.h>

// ---------------------------------------------------------------------------
// AdaptivePoolingAttention for gfx1250 (MI455X).
// bf16 WMMA pipeline + async global->LDS double-buffered GEMM staging +
// L2-supertile grid swizzle (192MB L2 keeps wkv(bf16) + A-strips resident).
// SEG=32, L=512, R=64, POOL=8, H=32, HD=128, D=4096.
// ---------------------------------------------------------------------------

typedef __bf16 bf16;
typedef __attribute__((ext_vector_type(16))) __bf16 v16bf;
typedef __attribute__((ext_vector_type(8)))  float  v8f;

// async builtin operand types: int4 in AS1 (global) / AS3 (LDS)
typedef int v4i_ __attribute__((vector_size(16)));
typedef __attribute__((address_space(1))) v4i_* as1_v4i;
typedef __attribute__((address_space(3))) v4i_* as3_v4i;

#define SEGN 32
#define LCTX 512
#define RQ   64
#define POOLN 8
#define NH   32
#define HDIM 128
#define DMODEL 4096
#define NKV  8192
#define MTOK (SEGN * LCTX)   // 16384 tokens
#define MQ   (SEGN * RQ)     // 2048 pooled queries
#define EPS  1e-5f
#define ATTN_SCALE 0.08838834764831843f  // 128^-0.5

// ---------------------------------------------------------------------------
// low-level helpers
// ---------------------------------------------------------------------------

__device__ inline v16bf frag_ld2(const bf16* p0, const bf16* p1) {
  union { v16bf v; uint4 u[2]; } f;
  f.u[0] = *reinterpret_cast<const uint4*>(p0);
  f.u[1] = *reinterpret_cast<const uint4*>(p1);
  return f.v;
}

__device__ inline v8f wmma_bf16(v16bf a, v16bf b, v8f c) {
  return __builtin_amdgcn_wmma_f32_16x16x32_bf16(
      /*neg_a=*/false, a, /*neg_b=*/false, b,
      /*c_mod=*/(short)0, c, /*reuse_a=*/false, /*reuse_b=*/false);
}

// async copy of 16B global -> LDS (GLOBAL_LOAD_ASYNC_TO_LDS_B128, ASYNCcnt)
__device__ inline void async_b128(const bf16* gp, bf16* lp) {
#if __has_builtin(__builtin_amdgcn_global_load_async_to_lds_b128)
  __builtin_amdgcn_global_load_async_to_lds_b128(
      (as1_v4i)(v4i_*)const_cast<bf16*>(gp),
      (as3_v4i)(v4i_*)lp, 0, 0);
#else
  asm volatile("global_load_async_to_lds_b128 %0, %1, off"
               :: "v"((unsigned)(unsigned long long)lp), "v"(gp)
               : "memory");
#endif
}

__device__ inline void wait_async_0() {
#if __has_builtin(__builtin_amdgcn_s_wait_asynccnt)
  __builtin_amdgcn_s_wait_asynccnt(0);
#else
  asm volatile("s_wait_asynccnt 0x0" ::: "memory");
#endif
}
__device__ inline void wait_async_4() {
#if __has_builtin(__builtin_amdgcn_s_wait_asynccnt)
  __builtin_amdgcn_s_wait_asynccnt(4);
#else
  asm volatile("s_wait_asynccnt 0x4" ::: "memory");
#endif
}

// stage a 128x32 bf16 tile (row stride gstride elems) into LDS via async DMA.
// 512 16B-chunks, 256 threads -> 2 async instructions per thread per tile.
__device__ inline void async_tile_128x32(const bf16* __restrict__ g, int gstride,
                                         bf16* lbase, int tid) {
#pragma unroll
  for (int i = 0; i < 2; ++i) {
    int c = tid + i * 256;
    int row = c >> 2;
    int seg = c & 3;
    async_b128(g + (size_t)row * gstride + seg * 8, lbase + row * 32 + seg * 8);
  }
}

__device__ inline float block_reduce_sum(float v, float* sbuf) {
#pragma unroll
  for (int off = 16; off > 0; off >>= 1) v += __shfl_xor(v, off, 32);
  int wave = threadIdx.x >> 5, lane = threadIdx.x & 31;
  if (lane == 0) sbuf[wave] = v;
  __syncthreads();
  float r = (threadIdx.x < 8) ? sbuf[threadIdx.x] : 0.0f;
  if (wave == 0) {
#pragma unroll
    for (int off = 4; off > 0; off >>= 1) r += __shfl_xor(r, off, 32);
    if (lane == 0) sbuf[0] = r;
  }
  __syncthreads();
  float out = sbuf[0];
  __syncthreads();
  return out;
}

// ---------------------------------------------------------------------------
// cast / transpose-cast preprocessing (one pass; makes every GEMM operand
// bf16 so async LDS staging needs no conversion and wkv fits in the 192MB L2)
// ---------------------------------------------------------------------------
__global__ __launch_bounds__(256) void cast_bf16_kernel(
    const float* __restrict__ s, bf16* __restrict__ d, int n4) {
  int i = blockIdx.x * 256 + threadIdx.x;
  if (i < n4) {
    float4 f = reinterpret_cast<const float4*>(s)[i];
    union { bf16 h[4]; uint2 u; } r;
    r.h[0] = (bf16)f.x; r.h[1] = (bf16)f.y;
    r.h[2] = (bf16)f.z; r.h[3] = (bf16)f.w;
    reinterpret_cast<uint2*>(d)[i] = r.u;
  }
}

// dst[N][K] (bf16) = transpose(src[K][N] f32); 32x32 LDS tiles
__global__ __launch_bounds__(256) void transpose_cast_kernel(
    const float* __restrict__ src, bf16* __restrict__ dst, int rows, int cols) {
  __shared__ bf16 tile[32][33];
  int bx = blockIdx.x * 32;  // col base (becomes output row)
  int by = blockIdx.y * 32;  // row base
  int tx = threadIdx.x & 31, ty = threadIdx.x >> 5;
#pragma unroll
  for (int i = 0; i < 32; i += 8)
    tile[ty + i][tx] = (bf16)src[(size_t)(by + ty + i) * cols + bx + tx];
  __syncthreads();
#pragma unroll
  for (int i = 0; i < 32; i += 8)
    dst[(size_t)(bx + ty + i) * rows + by + tx] = tile[tx][ty + i];
}

// ---------------------------------------------------------------------------
// Kernel: pooling (mean of 8 tokens) + rmsnorm -> queries(f32), qn(bf16)
// ---------------------------------------------------------------------------
__global__ __launch_bounds__(256) void pool_norm_kernel(
    const float* __restrict__ x, const float* __restrict__ w,
    float* __restrict__ queries, bf16* __restrict__ qn) {
  int row = blockIdx.x;  // 0..2047
  const float* xb = x + (size_t)row * POOLN * DMODEL;
  __shared__ float red[8];
  float vals[16];
  float ss = 0.0f;
#pragma unroll
  for (int j = 0; j < 16; ++j) {
    int c = threadIdx.x + j * 256;
    float s = 0.0f;
#pragma unroll
    for (int r = 0; r < POOLN; ++r) s += xb[(size_t)r * DMODEL + c];
    s *= (1.0f / POOLN);
    vals[j] = s;
    ss += s * s;
  }
  float tot = block_reduce_sum(ss, red);
  float rs = rsqrtf(tot / DMODEL + EPS);
#pragma unroll
  for (int j = 0; j < 16; ++j) {
    int c = threadIdx.x + j * 256;
    queries[(size_t)row * DMODEL + c] = vals[j];
    qn[(size_t)row * DMODEL + c] = (bf16)(vals[j] * rs * w[c]);
  }
}

// ---------------------------------------------------------------------------
// Generic bf16 WMMA GEMM: C = A(MxK,bf16 row-major) * BT(NxK,bf16 row-major)^T
// Double-buffered async global->LDS staging; L2 supertile swizzle (G=16
// block-rows per group so the A strip + all of B stay L2-resident).
//   MODE 0: C -> f32 row-major
//   MODE 1: kv split -> k[seg][h][l][hd] bf16, vT[seg][h][hd][l] bf16
//   MODE 2: q reshape -> q[seg][h][r][hd] bf16
// block tile 128x128, BK=32, 8 waves (2x4), wave tile 64x32
// ---------------------------------------------------------------------------
template <int MODE>
__global__ __launch_bounds__(256) void gemm_kernel(
    const bf16* __restrict__ A, const bf16* __restrict__ BT,
    void* __restrict__ C0, void* __restrict__ C1, int M, int N, int K) {
  __shared__ bf16 As[2][128 * 32];  // 2 x 8 KB
  __shared__ bf16 Bs[2][128 * 32];  // 2 x 8 KB

  const int tid = threadIdx.x;
  const int lane = tid & 31;
  const int wave = tid >> 5;

  // L2-friendly rasterization: groups of 16 block-rows sweep all block-cols
  const int nbn = N >> 7;
  const int grpSize = 16 * nbn;
  const int grp = blockIdx.x / grpSize;
  const int rem = blockIdx.x - grp * grpSize;
  const int row0 = (grp * 16 + (rem & 15)) * 128;
  const int col0 = (rem >> 4) * 128;

  const int wm = (wave & 1) * 64;
  const int wn = (wave >> 1) * 32;
  const int mrow = lane & 15;
  const int kg = lane >> 4;

  v8f acc[4][2];
#pragma unroll
  for (int mt = 0; mt < 4; ++mt)
#pragma unroll
    for (int nt = 0; nt < 2; ++nt) acc[mt][nt] = {};

  const bf16* Ablk = A + (size_t)row0 * K;
  const bf16* Bblk = BT + (size_t)col0 * K;
  const int nK = K >> 5;

  // prologue: stage tile 0
  async_tile_128x32(Ablk, K, As[0], tid);
  async_tile_128x32(Bblk, K, Bs[0], tid);

  for (int ks = 0; ks < nK; ++ks) {
    const int cur = ks & 1;
    const bool more = (ks + 1) < nK;
    if (more) {  // overlap: stage next tile while current computes
      async_tile_128x32(Ablk + (ks + 1) * 32, K, As[cur ^ 1], tid);
      async_tile_128x32(Bblk + (ks + 1) * 32, K, Bs[cur ^ 1], tid);
      wait_async_4();  // only next tile's 4 asyncs may remain in flight
    } else {
      wait_async_0();
    }
    __syncthreads();

    v16bf af[4], bfr[2];
#pragma unroll
    for (int mt = 0; mt < 4; ++mt) {
      const bf16* p = As[cur] + (wm + mt * 16 + mrow) * 32;
      af[mt] = frag_ld2(p + kg * 16, p + kg * 16 + 8);
    }
#pragma unroll
    for (int nt = 0; nt < 2; ++nt) {
      const bf16* p = Bs[cur] + (wn + nt * 16 + mrow) * 32 + kg * 16;
      bfr[nt] = frag_ld2(p, p + 8);
    }
#pragma unroll
    for (int mt = 0; mt < 4; ++mt)
#pragma unroll
      for (int nt = 0; nt < 2; ++nt)
        acc[mt][nt] = wmma_bf16(af[mt], bfr[nt], acc[mt][nt]);
    __syncthreads();  // reads done before next overwrite of this buffer
  }

  // ---- epilogue ----
#pragma unroll
  for (int mt = 0; mt < 4; ++mt) {
#pragma unroll
    for (int nt = 0; nt < 2; ++nt) {
      int col = col0 + wn + nt * 16 + (lane & 15);
#pragma unroll
      for (int i = 0; i < 8; ++i) {
        int row = row0 + wm + mt * 16 + (lane >> 4) * 8 + i;
        float val = acc[mt][nt][i];
        if (MODE == 0) {
          reinterpret_cast<float*>(C0)[(size_t)row * N + col] = val;
        } else if (MODE == 1) {
          int seg = row >> 9, l = row & 511;
          if (col < DMODEL) {
            int h = col >> 7, hd = col & 127;
            reinterpret_cast<bf16*>(C0)[(((size_t)(seg * NH + h)) * LCTX + l) * HDIM + hd] =
                (bf16)val;
          } else {
            int c2 = col - DMODEL;
            int h = c2 >> 7, hd = c2 & 127;
            reinterpret_cast<bf16*>(C1)[(((size_t)(seg * NH + h)) * HDIM + hd) * LCTX + l] =
                (bf16)val;
          }
        } else {  // MODE 2: q reshape [seg][h][r][hd]
          int seg = row >> 6, r = row & 63;
          int h = col >> 7, hd = col & 127;
          reinterpret_cast<bf16*>(C0)[(((size_t)(seg * NH + h)) * RQ + r) * HDIM + hd] =
              (bf16)val;
        }
      }
    }
  }
}

// ---------------------------------------------------------------------------
// Fused per-(segment,head) attention.
// q[64x128] -> LDS; scores = q k^T * scale (WMMA, f32 in LDS);
// softmax (f32); o = attn v (WMMA); write o row-major [2048][4096] bf16.
// LDS: 16KB q + 128KB scores + 64KB attn = 208KB dynamic (<= 320KB/WGP).
// ---------------------------------------------------------------------------
__global__ __launch_bounds__(256) void attn_kernel(
    const bf16* __restrict__ qb, const bf16* __restrict__ kb,
    const bf16* __restrict__ vTb, bf16* __restrict__ ob) {
  extern __shared__ char smem[];
  bf16*  qs     = reinterpret_cast<bf16*>(smem);                    // [64][128]
  float* scores = reinterpret_cast<float*>(smem + 16384);           // [64][512]
  bf16*  attnb  = reinterpret_cast<bf16*>(smem + 16384 + 131072);   // [64][512]

  const int sh = blockIdx.x;  // seg*32 + h
  const int seg = sh >> 5;
  const int h = sh & 31;
  const int tid = threadIdx.x;
  const int lane = tid & 31;
  const int wave = tid >> 5;
  const int mrow = lane & 15;
  const int kg = lane >> 4;

  const bf16* qsrc = qb + (size_t)sh * RQ * HDIM;     // [64][128]
  const bf16* ksrc = kb + (size_t)sh * LCTX * HDIM;   // [512][128]
  const bf16* vsrc = vTb + (size_t)sh * HDIM * LCTX;  // [128][512]

  // load q tile into LDS (1024 uint4)
  {
    const uint4* s = reinterpret_cast<const uint4*>(qsrc);
    uint4* d = reinterpret_cast<uint4*>(qs);
#pragma unroll
    for (int j = 0; j < 4; ++j) d[tid + j * 256] = s[tid + j * 256];
  }
  __syncthreads();

  // scores: wave w owns columns [w*64, w*64+64), full M=64, K=128
  {
    v8f sc[4][4];
#pragma unroll
    for (int mt = 0; mt < 4; ++mt)
#pragma unroll
      for (int nt = 0; nt < 4; ++nt) sc[mt][nt] = {};

    for (int ks = 0; ks < 4; ++ks) {
      int k0 = ks * 32;
      v16bf af[4], bfr[4];
#pragma unroll
      for (int mt = 0; mt < 4; ++mt) {
        const bf16* p = qs + (size_t)(mt * 16 + mrow) * HDIM + k0;
        af[mt] = frag_ld2(p + kg * 16, p + kg * 16 + 8);
      }
#pragma unroll
      for (int nt = 0; nt < 4; ++nt) {
        int n = wave * 64 + nt * 16 + mrow;                      // token col
        const bf16* p = ksrc + (size_t)n * HDIM + k0 + kg * 16;  // K-contig
        bfr[nt] = frag_ld2(p, p + 8);
      }
#pragma unroll
      for (int mt = 0; mt < 4; ++mt)
#pragma unroll
        for (int nt = 0; nt < 4; ++nt)
          sc[mt][nt] = wmma_bf16(af[mt], bfr[nt], sc[mt][nt]);
    }
#pragma unroll
    for (int mt = 0; mt < 4; ++mt)
#pragma unroll
      for (int nt = 0; nt < 4; ++nt) {
        int ncol = wave * 64 + nt * 16 + (lane & 15);
#pragma unroll
        for (int i = 0; i < 8; ++i) {
          int row = mt * 16 + (lane >> 4) * 8 + i;
          scores[(size_t)row * LCTX + ncol] = sc[mt][nt][i] * ATTN_SCALE;
        }
      }
  }
  __syncthreads();

  // softmax over L=512 per row; 4 lanes per row
  {
    int row = tid >> 2;
    int qq = tid & 3;
    float* srow = scores + (size_t)row * LCTX + qq * 128;
    float mx = -3.0e38f;
#pragma unroll 8
    for (int j = 0; j < 128; ++j) mx = fmaxf(mx, srow[j]);
    mx = fmaxf(mx, __shfl_xor(mx, 1, 32));
    mx = fmaxf(mx, __shfl_xor(mx, 2, 32));
    float sum = 0.0f;
#pragma unroll 8
    for (int j = 0; j < 128; ++j) {
      float e = __expf(srow[j] - mx);
      srow[j] = e;
      sum += e;
    }
    sum += __shfl_xor(sum, 1, 32);
    sum += __shfl_xor(sum, 2, 32);
    float inv = 1.0f / sum;
    bf16* arow = attnb + (size_t)row * LCTX + qq * 128;
#pragma unroll 8
    for (int j = 0; j < 128; ++j) arow[j] = (bf16)(srow[j] * inv);
  }
  __syncthreads();

  // o = attn @ v : M=64, N=128, K=512; wave owns (mt = w&3, 4 n-tiles)
  {
    const int mt = wave & 3;
    const int nbase = (wave >> 2) * 4;
    v8f oc[4];
#pragma unroll
    for (int nt = 0; nt < 4; ++nt) oc[nt] = {};

    for (int ks = 0; ks < 16; ++ks) {
      int k0 = ks * 32;
      const bf16* pa = attnb + (size_t)(mt * 16 + mrow) * LCTX + k0;
      v16bf af = frag_ld2(pa + kg * 16, pa + kg * 16 + 8);
#pragma unroll
      for (int nt = 0; nt < 4; ++nt) {
        int col = (nbase + nt) * 16 + mrow;                         // hd index
        const bf16* pb = vsrc + (size_t)col * LCTX + k0 + kg * 16;  // K-contig
        v16bf bfrg = frag_ld2(pb, pb + 8);
        oc[nt] = wmma_bf16(af, bfrg, oc[nt]);
      }
    }
#pragma unroll
    for (int nt = 0; nt < 4; ++nt) {
      int col = (nbase + nt) * 16 + (lane & 15);
#pragma unroll
      for (int i = 0; i < 8; ++i) {
        int row = mt * 16 + (lane >> 4) * 8 + i;
        ob[(size_t)(seg * RQ + row) * DMODEL + h * HDIM + col] = (bf16)oc[nt][i];
      }
    }
  }
}

// ---------------------------------------------------------------------------
// out = rmsnorm(out_pre + queries); one block per row
// ---------------------------------------------------------------------------
__global__ __launch_bounds__(256) void resid_norm_kernel(
    const float* __restrict__ pre, const float* __restrict__ queries,
    const float* __restrict__ w, float* __restrict__ out) {
  int row = blockIdx.x;
  __shared__ float red[8];
  float vals[16];
  float ss = 0.0f;
#pragma unroll
  for (int j = 0; j < 16; ++j) {
    int c = threadIdx.x + j * 256;
    float v = pre[(size_t)row * DMODEL + c] + queries[(size_t)row * DMODEL + c];
    vals[j] = v;
    ss += v * v;
  }
  float tot = block_reduce_sum(ss, red);
  float rs = rsqrtf(tot / DMODEL + EPS);
#pragma unroll
  for (int j = 0; j < 16; ++j) {
    int c = threadIdx.x + j * 256;
    out[(size_t)row * DMODEL + c] = vals[j] * rs * w[c];
  }
}

// ---------------------------------------------------------------------------
// launch
// ---------------------------------------------------------------------------
extern "C" void kernel_launch(void* const* d_in, const int* in_sizes, int n_in,
                              void* d_out, int out_size, void* d_ws, size_t ws_size,
                              hipStream_t stream) {
  const float* x           = (const float*)d_in[0];
  const float* attn_norm_w = (const float*)d_in[1];
  const float* wq          = (const float*)d_in[2];
  const float* wkv         = (const float*)d_in[3];
  const float* wo          = (const float*)d_in[4];
  const float* out_norm_w  = (const float*)d_in[5];
  float* out = (float*)d_out;

  // workspace carve-up (~620 MB)
  char* ws = (char*)d_ws;
  size_t off = 0;
  auto carve = [&](size_t bytes) {
    void* p = ws + off;
    off += (bytes + 255) & ~(size_t)255;
    return p;
  };
  bf16*  xb      = (bf16*)carve((size_t)MTOK * DMODEL * 2);      // 128 MB
  bf16*  wkvT    = (bf16*)carve((size_t)NKV * DMODEL * 2);       // 64 MB
  bf16*  wqT     = (bf16*)carve((size_t)DMODEL * DMODEL * 2);    // 32 MB
  bf16*  woT     = (bf16*)carve((size_t)DMODEL * DMODEL * 2);    // 32 MB
  bf16*  qn_b    = (bf16*)carve((size_t)MQ * DMODEL * 2);        // 16 MB
  float* queries = (float*)carve((size_t)MQ * DMODEL * 4);       // 32 MB
  bf16*  kb      = (bf16*)carve((size_t)MTOK * DMODEL * 2);      // 128 MB
  bf16*  vTb     = (bf16*)carve((size_t)MTOK * DMODEL * 2);      // 128 MB
  bf16*  qb      = (bf16*)carve((size_t)MQ * DMODEL * 2);        // 16 MB
  bf16*  ob      = (bf16*)carve((size_t)MQ * DMODEL * 2);        // 16 MB
  float* out_pre = (float*)xb;  // alias: xb is dead after the kv GEMM

  // 0. precision/layout prep: x->bf16, weights -> transposed bf16
  cast_bf16_kernel<<<(MTOK * DMODEL / 4 + 255) / 256, 256, 0, stream>>>(
      x, xb, MTOK * DMODEL / 4);
  transpose_cast_kernel<<<dim3(NKV / 32, DMODEL / 32), 256, 0, stream>>>(
      wkv, wkvT, DMODEL, NKV);
  transpose_cast_kernel<<<dim3(DMODEL / 32, DMODEL / 32), 256, 0, stream>>>(
      wq, wqT, DMODEL, DMODEL);
  transpose_cast_kernel<<<dim3(DMODEL / 32, DMODEL / 32), 256, 0, stream>>>(
      wo, woT, DMODEL, DMODEL);

  // 1. pooling + rmsnorm
  pool_norm_kernel<<<MQ, 256, 0, stream>>>(x, attn_norm_w, queries, qn_b);

  // 2. kv = x @ wkv  (16384x4096 * 4096x8192), split into k / vT bf16
  gemm_kernel<1><<<(MTOK / 128) * (NKV / 128), 256, 0, stream>>>(
      xb, wkvT, kb, vTb, MTOK, NKV, DMODEL);

  // 3. q = qn @ wq  (2048x4096 * 4096x4096), head-major reshape
  gemm_kernel<2><<<(MQ / 128) * (DMODEL / 128), 256, 0, stream>>>(
      qn_b, wqT, qb, nullptr, MQ, DMODEL, DMODEL);

  // 4. fused attention per (seg, head)
  const size_t attn_smem = 16384 + 131072 + 65536;  // 208 KB dynamic LDS
  (void)hipFuncSetAttribute(reinterpret_cast<const void*>(attn_kernel),
                            hipFuncAttributeMaxDynamicSharedMemorySize,
                            (int)attn_smem);
  attn_kernel<<<SEGN * NH, 256, attn_smem, stream>>>(qb, kb, vTb, ob);

  // 5. out_pre = o @ wo  (2048x4096 * 4096x4096), f32 out
  gemm_kernel<0><<<(MQ / 128) * (DMODEL / 128), 256, 0, stream>>>(
      ob, woT, out_pre, nullptr, MQ, DMODEL, DMODEL);

  // 6. residual + rmsnorm -> d_out
  resid_norm_kernel<<<MQ, 256, 0, stream>>>(out_pre, queries, out_norm_w, out);
}
